// RelaxTripletLoss_37984690766019
// MI455X (gfx1250) — compile-verified, matrix operand order checked by reading, not run on hardware
//
#include <hip/hip_runtime.h>
#include <hip/hip_bf16.h>

typedef __attribute__((ext_vector_type(16))) __bf16 v16bf;
typedef __attribute__((ext_vector_type(8)))  __bf16 v8bf;
typedef __attribute__((ext_vector_type(8)))  float  v8f;

#define MARGIN 0.3f

// ---------------------------------------------------------------------------
// Kernel 1: split fp32 -> bf16 hi + bf16 lo (x = hi + lo to ~bf16^2 precision)
// ---------------------------------------------------------------------------
__global__ void split_bf16_kernel(const float* __restrict__ x,
                                  __bf16* __restrict__ hi,
                                  __bf16* __restrict__ lo,
                                  size_t total) {
    size_t i = (size_t)blockIdx.x * blockDim.x + threadIdx.x;
    size_t stride = (size_t)gridDim.x * blockDim.x;
    for (; i < total; i += stride) {
        float v = x[i];
        __bf16 h = (__bf16)v;
        hi[i] = h;
        lo[i] = (__bf16)(v - (float)h);
    }
}

// ---------------------------------------------------------------------------
// Kernel 2: exact fp32 row squared norms. One block per row.
// ---------------------------------------------------------------------------
__global__ void rowsq_kernel(const float* __restrict__ x,
                             float* __restrict__ sq, int d) {
    __shared__ float red[256];
    const float* row = x + (size_t)blockIdx.x * d;
    float s = 0.f;
    for (int j = threadIdx.x; j < d; j += 256) {
        float v = row[j];
        s += v * v;
    }
    red[threadIdx.x] = s;
    __syncthreads();
    for (int o = 128; o > 0; o >>= 1) {
        if (threadIdx.x < o) red[threadIdx.x] += red[threadIdx.x + o];
        __syncthreads();
    }
    if (threadIdx.x == 0) sq[blockIdx.x] = red[0];
}

// ---------------------------------------------------------------------------
// Kernel 3: init reduction buffers + row-0 "has zero-class negative" gate.
// Single block (ordering-safe without grid sync).
// ---------------------------------------------------------------------------
__global__ void init_kernel(const int* __restrict__ targets,
                            unsigned* __restrict__ ap_bits,
                            unsigned* __restrict__ min_bits,
                            unsigned* __restrict__ flag, int n) {
    __shared__ unsigned sflag;
    if (threadIdx.x == 0) sflag = 0u;
    __syncthreads();
    int t0 = targets[0];
    unsigned local = 0u;
    for (int i = threadIdx.x; i < n; i += blockDim.x) {
        ap_bits[i]  = 0u;           // dist >= 0, uint-max == float-max
        min_bits[i] = 0x7F800000u;  // +inf
        int tj = targets[i];
        if ((tj != t0) && (t0 == 0 || tj == 0)) local = 1u;
    }
    if (local) atomicOr(&sflag, 1u);
    __syncthreads();
    if (threadIdx.x == 0) *flag = sflag;
}

// ---------------------------------------------------------------------------
// WMMA fragment load: 16-bit A/B layout (ISA 7.12.2).
// Lane L<16: row r, K = {0..7, 16..23}; lane L>=16: row r, K = {8..15, 24..31}.
// p points at (row, kc + sel*8); second 128b chunk is +16 elements (+32 B).
// ---------------------------------------------------------------------------
static __device__ inline v16bf load_frag(const __bf16* p) {
    v8bf c0 = *(const v8bf*)(p);
    v8bf c1 = *(const v8bf*)(p + 16);
    v16bf r;
#pragma unroll
    for (int i = 0; i < 8; ++i) { r[i] = c0[i]; r[i + 8] = c1[i]; }
    return r;
}

static __device__ inline v8f wmma_bf16(v16bf a, v16bf b, v8f c) {
    return __builtin_amdgcn_wmma_f32_16x16x32_bf16(
        false, a, false, b, (short)0, c, false, false);
}

// ---------------------------------------------------------------------------
// Kernel 4: Gram tile + fused distance + masked row max/min reductions.
// Block tile 128(M) x 128(N); 8 waves arranged 2(M) x 4(N); wave tile 64x32.
// Split-bf16: G ~= hi*hi + hi*lo + lo*hi accumulated in fp32.
//
// Addressing: keep only the addrspace(1) kernel-arg bases (xhi/xlo) and carry
// SIX 32-bit element offsets (shared between hi and lo). With the 8-step
// inner loop fully unrolled, every load is SADDR-base + 32-bit voffset +
// constant immediate -> global_load_b128, zero address VALU in the body,
// no FLAT (no DScnt coupling), no WMMA->VALU hazard NOPs.
// ---------------------------------------------------------------------------
#define KBLK 256   // K elements per outer iteration (8 sub-steps of 32)

__global__ __launch_bounds__(256)
void gram_kernel(const __bf16* __restrict__ xhi,
                 const __bf16* __restrict__ xlo,
                 const float* __restrict__ sq,
                 const int* __restrict__ targets,
                 unsigned* __restrict__ ap_bits,
                 unsigned* __restrict__ min_bits,
                 int n, int d) {
    const int lane  = threadIdx.x & 31;
    const int wave  = threadIdx.x >> 5;
    const int waveM = wave >> 2;   // 0..1
    const int waveN = wave & 3;    // 0..3
    const int m0 = blockIdx.y * 128 + waveM * 64;
    const int n0 = blockIdx.x * 128 + waveN * 32;
    const int rlane = lane & 15;
    const int sel   = lane >> 4;

    // 32-bit element offsets of each fragment row (shared by xhi and xlo).
    unsigned offA[4], offB[2];
#pragma unroll
    for (int mi = 0; mi < 4; ++mi)
        offA[mi] = (unsigned)((m0 + mi * 16 + rlane) * d + sel * 8);
#pragma unroll
    for (int ni = 0; ni < 2; ++ni)
        offB[ni] = (unsigned)((n0 + ni * 16 + rlane) * d + sel * 8);

    v8f acc[4][2] = {};

    for (int ko = 0; ko < d; ko += KBLK) {
        // Prefetch next K block into cache (global_prefetch_b8).
        if (ko + KBLK < d) {
#pragma unroll
            for (int mi = 0; mi < 4; ++mi) {
                __builtin_prefetch(xhi + offA[mi] + KBLK, 0, 3);
                __builtin_prefetch(xlo + offA[mi] + KBLK, 0, 3);
            }
#pragma unroll
            for (int ni = 0; ni < 2; ++ni) {
                __builtin_prefetch(xhi + offB[ni] + KBLK, 0, 3);
                __builtin_prefetch(xlo + offB[ni] + KBLK, 0, 3);
            }
        }
        // 8 K-steps with constant immediate offsets: zero address math here.
#pragma unroll
        for (int ks = 0; ks < KBLK; ks += 32) {
            v16bf ah[4], al[4], bh[2], bl[2];
#pragma unroll
            for (int mi = 0; mi < 4; ++mi) {
                ah[mi] = load_frag(xhi + offA[mi] + ks);
                al[mi] = load_frag(xlo + offA[mi] + ks);
            }
#pragma unroll
            for (int ni = 0; ni < 2; ++ni) {
                bh[ni] = load_frag(xhi + offB[ni] + ks);
                bl[ni] = load_frag(xlo + offB[ni] + ks);
            }
#pragma unroll
            for (int mi = 0; mi < 4; ++mi)
#pragma unroll
                for (int ni = 0; ni < 2; ++ni) {
                    acc[mi][ni] = wmma_bf16(ah[mi], bh[ni], acc[mi][ni]);
                    acc[mi][ni] = wmma_bf16(ah[mi], bl[ni], acc[mi][ni]);
                    acc[mi][ni] = wmma_bf16(al[mi], bh[ni], acc[mi][ni]);
                }
        }
        // Six 32-bit offset bumps per 192 WMMAs.
#pragma unroll
        for (int mi = 0; mi < 4; ++mi) offA[mi] += KBLK;
#pragma unroll
        for (int ni = 0; ni < 2; ++ni) offB[ni] += KBLK;
    }

    // Epilogue. C/D layout: VGPR e, lanes 0-15 -> (M=e, N=lane),
    // lanes 16-31 -> (M=e+8, N=lane-16).
#pragma unroll
    for (int mi = 0; mi < 4; ++mi) {
#pragma unroll
        for (int e = 0; e < 8; ++e) {
            int m = m0 + mi * 16 + e + sel * 8;
            float sqm = sq[m];
            int   tm  = targets[m];
            float rmax = 0.0f;                        // identity for nonneg max
            float rmin = __builtin_inff();            // identity for min
#pragma unroll
            for (int ni = 0; ni < 2; ++ni) {
                int nn = n0 + ni * 16 + rlane;
                float g  = acc[mi][ni][e];
                float d2 = sqm + sq[nn] - 2.0f * g;
                float dist = __builtin_sqrtf(fmaxf(d2, 1e-12f));
                int tn = targets[nn];
                bool same = (tm == tn);
                bool zn   = (!same) && (tm == 0 || tn == 0);
                if (same) rmax = fmaxf(rmax, dist);
                if (zn)   rmin = fminf(rmin, dist);
            }
            // reduce across the 16-lane half (xor 1,2,4,8 stays in-half)
#pragma unroll
            for (int off = 1; off < 16; off <<= 1) {
                rmax = fmaxf(rmax, __shfl_xor(rmax, off));
                rmin = fminf(rmin, __shfl_xor(rmin, off));
            }
            if (rlane == 0) {
                atomicMax(&ap_bits[m],  __float_as_uint(rmax));
                atomicMin(&min_bits[m], __float_as_uint(rmin));
            }
        }
    }
}

// ---------------------------------------------------------------------------
// Kernel 5: loss = mean(relu(dist_ap - dist_an + margin)). Single block.
// ---------------------------------------------------------------------------
__global__ void final_kernel(const unsigned* __restrict__ ap_bits,
                             const unsigned* __restrict__ min_bits,
                             const unsigned* __restrict__ flag,
                             float* __restrict__ out, int n) {
    __shared__ float red[256];
    bool hz = (*flag) != 0u;
    float s = 0.f;
    for (int i = threadIdx.x; i < n; i += blockDim.x) {
        float ap = __uint_as_float(ap_bits[i]);
        float an = hz ? __uint_as_float(min_bits[i]) : 0.f;
        float t = ap - an + MARGIN;        // an may be +inf -> t=-inf -> relu 0
        s += fmaxf(t, 0.f);
    }
    red[threadIdx.x] = s;
    __syncthreads();
    for (int o = 128; o > 0; o >>= 1) {
        if (threadIdx.x < o) red[threadIdx.x] += red[threadIdx.x + o];
        __syncthreads();
    }
    if (threadIdx.x == 0) out[0] = red[0] / (float)n;
}

// ---------------------------------------------------------------------------
extern "C" void kernel_launch(void* const* d_in, const int* in_sizes, int n_in,
                              void* d_out, int out_size, void* d_ws, size_t ws_size,
                              hipStream_t stream) {
    const float* x       = (const float*)d_in[0];
    const int*   targets = (const int*)d_in[1];
    float*       out     = (float*)d_out;

    const int n = in_sizes[1];              // 4096
    const int d = in_sizes[0] / n;          // 2048
    const size_t total = (size_t)n * d;

    // workspace layout (all 16B aligned)
    char* ws = (char*)d_ws;
    const size_t xb = total * sizeof(__bf16);       // 16 MB each
    __bf16*   xhi      = (__bf16*)(ws);
    __bf16*   xlo      = (__bf16*)(ws + xb);
    float*    sq       = (float*)(ws + 2 * xb);
    unsigned* ap_bits  = (unsigned*)(ws + 2 * xb + (size_t)n * 4);
    unsigned* min_bits = (unsigned*)(ws + 2 * xb + (size_t)n * 8);
    unsigned* flag     = (unsigned*)(ws + 2 * xb + (size_t)n * 12);

    split_bf16_kernel<<<2048, 256, 0, stream>>>(x, xhi, xlo, total);
    rowsq_kernel<<<n, 256, 0, stream>>>(x, sq, d);
    init_kernel<<<1, 256, 0, stream>>>(targets, ap_bits, min_bits, flag, n);

    dim3 grid(n / 128, n / 128);
    gram_kernel<<<grid, 256, 0, stream>>>(xhi, xlo, sq, targets,
                                          ap_bits, min_bits, n, d);

    final_kernel<<<1, 256, 0, stream>>>(ap_bits, min_bits, flag, out, n);
}
